// long_window_LSTM_recurrentDropout_jit_25701084299313
// MI455X (gfx1250) — compile-verified, matrix operand order checked by reading, not run on hardware
//
#include <hip/hip_runtime.h>
#include <hip/hip_bf16.h>
#include <math.h>

// Problem constants (match reference)
constexpr int B   = 128;
constexpr int T   = 1024;
constexpr int IN  = 17;
constexpr int H   = 512;
constexpr int G4  = 4 * H;   // 2048
constexpr int OUT = 17;

constexpr int NBLK = 128;    // persistent workgroups: 4 M-slices x 32 col-slices
constexpr int NTHR = 256;    // 8 waves (wave32)

typedef __attribute__((ext_vector_type(16))) __bf16 v16bf;
typedef __attribute__((ext_vector_type(8)))  __bf16 v8bf;
typedef __attribute__((ext_vector_type(8)))  float  v8f;

// ---------------------------------------------------------------------------
// Fragment load matching the ISA 16-bit A/B VGPR layout:
//   lanes 0-15 : K = k..k+7  and K = k+16..k+23
//   lanes 16-31: K = k+8..k+15 and K = k+24..k+31
// Two contiguous 16-byte chunks -> two global_load_b128.
// Caller passes base = row_ptr + k + (lane>>4)*8.
// ---------------------------------------------------------------------------
__device__ __forceinline__ v16bf load_frag(const __bf16* base) {
  union { v16bf v; v8bf h[2]; } u;
  u.h[0] = *(const v8bf*)(base);
  u.h[1] = *(const v8bf*)(base + 16);
  return u.v;
}

// gfx1250 has a hardware v_tanh_f32 transcendental; use it for all LSTM
// activations (sigmoid via tanh identity). Guarded so compile never breaks.
__device__ __forceinline__ float fast_tanh(float x) {
#if __has_builtin(__builtin_amdgcn_tanhf)
  return __builtin_amdgcn_tanhf(x);
#elif __has_builtin(__builtin_amdgcn_tanh_f32)
  return __builtin_amdgcn_tanh_f32(x);
#else
  return tanhf(x);
#endif
}
__device__ __forceinline__ float sigm(float x) {
  return 0.5f * fast_tanh(0.5f * x) + 0.5f;
}

// ---------------------------------------------------------------------------
// Split grid barrier: per-class monotonic counters (classes: A = h0 ready,
// B = h1 ready). Safe because within a class a block cannot arrive at
// barrier k+1 before every block arrived at barrier k (its arrive k+1 is
// program-ordered after its wait k).
// ---------------------------------------------------------------------------
__device__ __forceinline__ void bar_arrive(int* bar) {
  __threadfence();            // every thread releases its h writes (agent/L2)
  __syncthreads();            // all threads' fences done before signal
  if (threadIdx.x == 0) {
    __hip_atomic_fetch_add(bar, 1, __ATOMIC_RELEASE, __HIP_MEMORY_SCOPE_AGENT);
  }
}
__device__ __forceinline__ void bar_wait(int* bar, int target) {
  if (threadIdx.x == 0) {
    while (__hip_atomic_load(bar, __ATOMIC_ACQUIRE, __HIP_MEMORY_SCOPE_AGENT) < target) {
      __builtin_amdgcn_s_sleep(1);
    }
  }
  __syncthreads();
  __threadfence();            // acquire: subsequent loads see fresh L2 data
}

// ---------------------------------------------------------------------------
// Prep: convert recurrent/input weight matrices to bf16, zero h buffers,
// reset barrier counters. Runs every call (deterministic, graph-safe).
// ---------------------------------------------------------------------------
__global__ void lstm_prep_kernel(const float* __restrict__ whh0,
                                 const float* __restrict__ wih1,
                                 const float* __restrict__ whh1,
                                 __bf16* __restrict__ whh0b,
                                 __bf16* __restrict__ wih1b,
                                 __bf16* __restrict__ whh1b,
                                 __bf16* __restrict__ h0buf,
                                 __bf16* __restrict__ h1buf,
                                 int* __restrict__ barA,
                                 int* __restrict__ barB) {
  const int stride = gridDim.x * blockDim.x;
  const int tid = blockIdx.x * blockDim.x + threadIdx.x;
  const int NW = G4 * H;
  for (int i = tid; i < NW; i += stride) {
    whh0b[i] = (__bf16)whh0[i];
    wih1b[i] = (__bf16)wih1[i];
    whh1b[i] = (__bf16)whh1[i];
  }
  const int NH = 2 * B * H;
  for (int i = tid; i < NH; i += stride) {
    h0buf[i] = (__bf16)0.0f;
    h1buf[i] = (__bf16)0.0f;
  }
  if (tid == 0) { *barA = 0; *barB = 0; }
}

// ---------------------------------------------------------------------------
// Persistent fused 2-layer LSTM scan with latency-hidden split barriers.
// block b: m_base = (b>>5)*32 (batch rows), col_base = (b&31)*16 (h columns)
// wave w : m_tile = w>>2 (0/1), gate g = w&3
// Per step, per wave: one 16x16 gate tile per layer via bf16 WMMA.
//
// Pipeline per step t:
//   L0 gates+update -> write h0_t -> arrive(A_t)
//   wait(B_{t-1})                      [hidden behind whole L0 phase]
//   L1 recurrent GEMM (h1_{t-1})       [independent of A_t]
//   wait(A_t)                          [hidden behind L1 recurrent GEMM]
//   L1 input GEMM (h0_t) + update -> write h1_t -> arrive(B_t)
// ---------------------------------------------------------------------------
__global__ void __launch_bounds__(NTHR)
lstm_scan_kernel(const float* __restrict__ x,
                 const float* __restrict__ wih0,
                 const float* __restrict__ bih0,
                 const float* __restrict__ bhh0,
                 const float* __restrict__ bih1,
                 const float* __restrict__ bhh1,
                 const __bf16* __restrict__ whh0b,
                 const __bf16* __restrict__ wih1b,
                 const __bf16* __restrict__ whh1b,
                 __bf16* __restrict__ h0buf,   // [2][B][H]
                 __bf16* __restrict__ h1buf,   // [2][B][H]
                 float*  __restrict__ h1f,     // [B][H] final h1 (fp32)
                 int*    __restrict__ barA,
                 int*    __restrict__ barB) {
  __shared__ float gates_s[2][4][16][16];  // [m_tile][gate][m_in_tile][n]
  __shared__ float xs[32 * IN];            // x_t rows for this block
  __shared__ float wih0_s[64 * IN];        // wih0 rows for this block's gate cols

  const int tid    = threadIdx.x;
  const int lane   = tid & 31;
  const int wave   = tid >> 5;
  const int mtile  = wave >> 2;            // 0..1
  const int g      = wave & 3;             // gate index
  const int n      = lane & 15;            // N within tile / frag row
  const int half   = lane >> 4;            // K-half selector

  const int m_base   = (blockIdx.x >> 5) * 32;
  const int col_base = (blockIdx.x & 31) * 16;

  // Preload this block's wih0 slice (64 gate-columns x 17) into LDS.
  for (int i = tid; i < 64 * IN; i += NTHR) {
    const int row = i / IN;                         // 0..63 -> gate*16 + n
    const int gcolr = (row >> 4) * H + col_base + (row & 15);
    wih0_s[i] = wih0[gcolr * IN + (i % IN)];
  }

  // Per-wave-lane gate column and fused biases.
  const int gcol = g * H + col_base + n;
  const float bias0v = bih0[gcol] + bhh0[gcol];
  const float bias1v = bih1[gcol] + bhh1[gcol];

  // A-fragment batch row and K sub-offset.
  const int am   = m_base + mtile * 16 + n;
  const int koff = half * 8;

  // B fragment row pointers (weight row == gate column; contiguous K).
  const __bf16* b0row = whh0b + (size_t)gcol * H + koff;
  const __bf16* b1row = wih1b + (size_t)gcol * H + koff;
  const __bf16* b2row = whh1b + (size_t)gcol * H + koff;

  // Per-thread cell state: 2 (row,col) pairs per layer, fixed mapping.
  float c0s[2] = {0.0f, 0.0f};
  float c1s[2] = {0.0f, 0.0f};

  for (int t = 0; t < T; ++t) {
    const int p_rd = t & 1;
    const int p_wr = (t + 1) & 1;

    // Stage x_t rows [m_base, m_base+32) into LDS.
    __syncthreads();  // previous consumers of xs are done
    for (int i = tid; i < 32 * IN; i += NTHR) {
      const int row = i / IN;
      xs[i] = x[(size_t)(m_base + row) * T * IN + (size_t)t * IN + (i % IN)];
    }
    __syncthreads();

    // ---------------- Layer 0 gate tile ----------------
    v8f acc;
#pragma unroll
    for (int r = 0; r < 8; ++r) acc[r] = bias0v;

    // Tiny K=17 input projection (VALU), reads from LDS.
    const int mrow = mtile * 16;
#pragma unroll 1
    for (int d = 0; d < IN; ++d) {
      const float wv = wih0_s[(g * 16 + n) * IN + d];
#pragma unroll
      for (int r = 0; r < 8; ++r)
        acc[r] += xs[(mrow + r + 8 * half) * IN + d] * wv;
    }

    // Recurrent GEMM: h0_{t-1} x whh0^T, bf16 WMMA. (h0_{t-1} synced A_{t-1})
    {
      const __bf16* a0row = h0buf + (size_t)p_rd * B * H + (size_t)am * H + koff;
#pragma unroll 4
      for (int k = 0; k < H; k += 32) {
        v16bf a  = load_frag(a0row + k);
        v16bf bb = load_frag(b0row + k);
        acc = __builtin_amdgcn_wmma_f32_16x16x32_bf16(
            false, a, false, bb, (short)0, acc, false, false);
      }
    }

#pragma unroll
    for (int r = 0; r < 8; ++r) gates_s[mtile][g][r + 8 * half][n] = acc[r];
    __syncthreads();

    // Layer-0 elementwise: each thread owns 2 (m,j) cells.
    {
      __bf16* h0n = h0buf + (size_t)p_wr * B * H;
#pragma unroll
      for (int pp = 0; pp < 2; ++pp) {
        const int p  = tid + NTHR * pp;
        const int ml = p >> 4, j = p & 15;
        const int mt = ml >> 4, mi = ml & 15;
        const float iv = sigm(gates_s[mt][0][mi][j]);
        const float fv = sigm(gates_s[mt][1][mi][j]);
        const float gv = fast_tanh(gates_s[mt][2][mi][j]);
        const float ov = sigm(gates_s[mt][3][mi][j]);
        const float cn = fv * c0s[pp] + iv * gv;
        c0s[pp] = cn;
        const float hv = ov * fast_tanh(cn);
        h0n[(size_t)(m_base + ml) * H + col_base + j] = (__bf16)hv;
      }
    }

    bar_arrive(barA);                 // A_t : h0_t published

    // wait for h1_{t-1} (B_{t-1}); latency hidden behind the whole L0 phase.
    bar_wait(barB, t * NBLK);

    // ---------------- Layer 1: recurrent GEMM first (independent of A_t) ---
    v8f acc1;
#pragma unroll
    for (int r = 0; r < 8; ++r) acc1[r] = bias1v;
    {
      const __bf16* a2row = h1buf + (size_t)p_rd * B * H + (size_t)am * H + koff;
#pragma unroll 4
      for (int k = 0; k < H; k += 32) {
        v16bf a  = load_frag(a2row + k);
        v16bf bb = load_frag(b2row + k);
        acc1 = __builtin_amdgcn_wmma_f32_16x16x32_bf16(
            false, a, false, bb, (short)0, acc1, false, false);
      }
    }

    // wait for h0_t (A_t); latency hidden behind the recurrent GEMM above.
    bar_wait(barA, (t + 1) * NBLK);

    // Input GEMM over fresh h0_t.
    {
      const __bf16* a1row = h0buf + (size_t)p_wr * B * H + (size_t)am * H + koff;
#pragma unroll 4
      for (int k = 0; k < H; k += 32) {
        v16bf a  = load_frag(a1row + k);
        v16bf bb = load_frag(b1row + k);
        acc1 = __builtin_amdgcn_wmma_f32_16x16x32_bf16(
            false, a, false, bb, (short)0, acc1, false, false);
      }
    }

#pragma unroll
    for (int r = 0; r < 8; ++r) gates_s[mtile][g][r + 8 * half][n] = acc1[r];
    __syncthreads();

    // Layer-1 elementwise.
    {
      __bf16* h1n = h1buf + (size_t)p_wr * B * H;
#pragma unroll
      for (int pp = 0; pp < 2; ++pp) {
        const int p  = tid + NTHR * pp;
        const int ml = p >> 4, j = p & 15;
        const int mt = ml >> 4, mi = ml & 15;
        const float iv = sigm(gates_s[mt][0][mi][j]);
        const float fv = sigm(gates_s[mt][1][mi][j]);
        const float gv = fast_tanh(gates_s[mt][2][mi][j]);
        const float ov = sigm(gates_s[mt][3][mi][j]);
        const float cn = fv * c1s[pp] + iv * gv;
        c1s[pp] = cn;
        const float hv = ov * fast_tanh(cn);
        h1n[(size_t)(m_base + ml) * H + col_base + j] = (__bf16)hv;
        if (t == T - 1)
          h1f[(size_t)(m_base + ml) * H + col_base + j] = hv;
      }
    }

    bar_arrive(barB);                 // B_t : h1_t published
  }
}

// ---------------------------------------------------------------------------
// Final linear: out[b,o] = lin_b[o] + h1_last[b,:] . lin_w[o,:]
// ---------------------------------------------------------------------------
__global__ void final_linear_kernel(const float* __restrict__ h1f,
                                    const float* __restrict__ lin_w,
                                    const float* __restrict__ lin_b,
                                    float* __restrict__ out) {
  const int t = blockIdx.x * blockDim.x + threadIdx.x;
  if (t >= B * OUT) return;
  const int b_ = t / OUT, o = t % OUT;
  float s = lin_b[o];
  const float* hr = h1f + (size_t)b_ * H;
  const float* wr = lin_w + (size_t)o * H;
#pragma unroll 8
  for (int k = 0; k < H; ++k) s += hr[k] * wr[k];
  out[t] = s;
}

// ---------------------------------------------------------------------------
// Launch. Inputs (setup_inputs order):
// 0:x 1:wih0 2:whh0 3:bih0 4:bhh0 5:wih1 6:whh1 7:bih1 8:bhh1 9:lin_w 10:lin_b
// Workspace (~6.9 MB): barriers, bf16 weights, double-buffered h, final h1.
// ---------------------------------------------------------------------------
extern "C" void kernel_launch(void* const* d_in, const int* in_sizes, int n_in,
                              void* d_out, int out_size, void* d_ws, size_t ws_size,
                              hipStream_t stream) {
  const float* x     = (const float*)d_in[0];
  const float* wih0  = (const float*)d_in[1];
  const float* whh0  = (const float*)d_in[2];
  const float* bih0  = (const float*)d_in[3];
  const float* bhh0  = (const float*)d_in[4];
  const float* wih1  = (const float*)d_in[5];
  const float* whh1  = (const float*)d_in[6];
  const float* bih1  = (const float*)d_in[7];
  const float* bhh1  = (const float*)d_in[8];
  const float* lin_w = (const float*)d_in[9];
  const float* lin_b = (const float*)d_in[10];
  float* out = (float*)d_out;

  char* w = (char*)d_ws;
  int*    barA  = (int*)w;                               // offset 0
  int*    barB  = (int*)(w + 128);                       // separate cacheline
  __bf16* whh0b = (__bf16*)(w + 256);
  __bf16* wih1b = whh0b + (size_t)G4 * H;
  __bf16* whh1b = wih1b + (size_t)G4 * H;
  __bf16* h0buf = whh1b + (size_t)G4 * H;                // [2][B][H]
  __bf16* h1buf = h0buf + (size_t)2 * B * H;             // [2][B][H]
  float*  h1f   = (float*)(h1buf + (size_t)2 * B * H);   // [B][H]

  lstm_prep_kernel<<<512, 256, 0, stream>>>(whh0, wih1, whh1,
                                            whh0b, wih1b, whh1b,
                                            h0buf, h1buf, barA, barB);

  lstm_scan_kernel<<<NBLK, NTHR, 0, stream>>>(x, wih0, bih0, bhh0, bih1, bhh1,
                                              whh0b, wih1b, whh1b,
                                              h0buf, h1buf, h1f, barA, barB);

  final_linear_kernel<<<(B * OUT + 255) / 256, 256, 0, stream>>>(h1f, lin_w,
                                                                 lin_b, out);
}
